// PDAN_63041529971189
// MI455X (gfx1250) — matrix operand order repaired and checked by reading, not compiled, not run on hardware
//
#include <hip/hip_runtime.h>

#define LAYERS 5
#define CCH    512
#define DIMX   1024
#define NCLS   157
#define NCPAD  160
#define BATCH  2
#define TLEN   4096

typedef __attribute__((ext_vector_type(16))) __bf16 v16bf;
typedef __attribute__((ext_vector_type(8)))  __bf16 v8bf;
typedef __attribute__((ext_vector_type(8)))  float  v8f;

__device__ __forceinline__ __bf16 f2bf(float f) {
  unsigned u = __float_as_uint(f);
  u += 0x7fffu + ((u >> 16) & 1u);            // round-to-nearest-even
  unsigned short s = (unsigned short)(u >> 16);
  __bf16 b;
  __builtin_memcpy(&b, &s, 2);
  return b;
}
__device__ __forceinline__ float bf2f(__bf16 b) {
  unsigned short s;
  __builtin_memcpy(&s, &b, 2);
  return __uint_as_float(((unsigned)s) << 16);
}

// ---------------------------------------------------------------------------
// WMMA GEMM:  out[b, o, t] = sum_c A[o, c] * Bmat[b, t, c]   (bf16 in, f32 acc)
//   A    : (M, K) row-major bf16 (weights)
//   Bmat : (BATCH, N, K) bf16, K contiguous (activations in (T,C) layout)
//   EPI 0: store bf16 (T,M) -> Obf
//   EPI 1: +bias, store bf16          (input projection)
//   EPI 2: +bias, h=(h+d)*mask, store bf16 in place (w1 + residual)
//   EPI 3: +bias, *mask, store f32 (B, validM, N) to Of32 (classifier head)
// Each wave: 32(M) x 64(N) tile, K in steps of 32 -> 8 wmma per step.
// ---------------------------------------------------------------------------
template<int EPI>
__global__ __launch_bounds__(256)
void wmma_gemm_kernel(const __bf16* __restrict__ A,
                      const __bf16* __restrict__ Bmat,
                      __bf16* __restrict__ Obf,
                      float* __restrict__ Of32,
                      const float* __restrict__ bias,
                      const float* __restrict__ mask,
                      int M, int N, int K, int validM)
{
  const int lane   = threadIdx.x & 31;
  const int wave   = (blockIdx.x * blockDim.x + threadIdx.x) >> 5;
  const int mTiles = M >> 5;
  const int nTiles = N >> 6;
  const int perB   = mTiles * nTiles;
  const int b      = wave / perB;
  const int rem    = wave % perB;
  const int o0     = (rem / nTiles) << 5;
  const int t0     = (rem % nTiles) << 6;
  const int lh     = lane >> 4;   // 0/1 half-wave
  const int ll     = lane & 15;

  v8f acc[2][4] = {};

  // A fragment base: lane ll -> row, 16 contiguous K at k+lh*16
  const __bf16* aBase0 = A + (size_t)(o0 + ll) * K + lh * 16;
  const __bf16* aBase1 = A + (size_t)(o0 + 16 + ll) * K + lh * 16;
  // B fragment base: lane ll -> column t, 16 contiguous K at k+lh*16
  const __bf16* bBase  = Bmat + ((size_t)b * N + t0 + ll) * K + lh * 16;

  for (int kk = 0; kk < K; kk += 32) {
    v16bf af[2];
    {
      v8bf lo0 = *(const v8bf*)(aBase0 + kk);
      v8bf hi0 = *(const v8bf*)(aBase0 + kk + 8);
      v8bf lo1 = *(const v8bf*)(aBase1 + kk);
      v8bf hi1 = *(const v8bf*)(aBase1 + kk + 8);
#pragma unroll
      for (int r = 0; r < 8; ++r) {
        af[0][r] = lo0[r]; af[0][8 + r] = hi0[r];
        af[1][r] = lo1[r]; af[1][8 + r] = hi1[r];
      }
    }
    v16bf bfv[4];
#pragma unroll
    for (int j = 0; j < 4; ++j)
      bfv[j] = *(const v16bf*)(bBase + (size_t)(16 * j) * K + kk);

    if (kk + 32 < K) {
      __builtin_prefetch(bBase + kk + 32, 0, 1);              // global_prefetch_b8
      __builtin_prefetch(bBase + (size_t)48 * K + kk + 32, 0, 1);
    }

#pragma unroll
    for (int i = 0; i < 2; ++i)
#pragma unroll
      for (int j = 0; j < 4; ++j)
        acc[i][j] = __builtin_amdgcn_wmma_f32_16x16x32_bf16(
            false, af[i], false, bfv[j], (short)0, acc[i][j], false, false);
  }

#pragma unroll
  for (int i = 0; i < 2; ++i) {
    const int ob = o0 + 16 * i + lh * 8;   // 8 consecutive output rows per lane
#pragma unroll
    for (int j = 0; j < 4; ++j) {
      const int t = t0 + 16 * j + ll;
      v8f d = acc[i][j];
      if (EPI == 1 || EPI == 2) {
#pragma unroll
        for (int r = 0; r < 8; ++r) d[r] += bias[ob + r];
      }
      if (EPI == 2) {
        const float mk = mask[(size_t)b * N + t];
        v8bf hold = *(const v8bf*)(Obf + ((size_t)b * N + t) * M + ob);
#pragma unroll
        for (int r = 0; r < 8; ++r) d[r] = (bf2f(hold[r]) + d[r]) * mk;
      }
      if (EPI == 3) {
        const float mk = mask[(size_t)b * N + t];
#pragma unroll
        for (int r = 0; r < 8; ++r) {
          const int nc = ob + r;
          if (nc < validM)
            Of32[((size_t)b * validM + nc) * N + t] = (d[r] + bias[nc]) * mk;
        }
      } else {
        v8bf s;
#pragma unroll
        for (int r = 0; r < 8; ++r) s[r] = f2bf(d[r]);
        *(v8bf*)(Obf + ((size_t)b * N + t) * M + ob) = s;
      }
    }
  }
}

// ---------------------------------------------------------------------------
// Dilated 3-tap attention + relu, all in (B,T,C) layout, 8 channels per thread.
// Out-of-range taps: k=0 (score q*0=0), v=0 (so vw = rel) -- matches zero-pad.
// ---------------------------------------------------------------------------
__global__ __launch_bounds__(256)
void dal_attn_kernel(const __bf16* __restrict__ q, const __bf16* __restrict__ k,
                     const __bf16* __restrict__ v, const float* __restrict__ rel,
                     __bf16* __restrict__ a, int d)
{
  const size_t idx = (size_t)blockIdx.x * blockDim.x + threadIdx.x;
  const int    cg  = (int)(idx % (CCH / 8));
  const size_t bt  = idx / (CCH / 8);
  if (bt >= (size_t)BATCH * TLEN) return;
  const int t  = (int)(bt % TLEN);
  const int b  = (int)(bt / TLEN);
  const int c0 = cg * 8;
  const size_t base = bt * CCH + c0;

  v8bf qv = *(const v8bf*)(q + base);
  v8bf kv[3], vv[3];
#pragma unroll
  for (int j = 0; j < 3; ++j) {
    const int tj = t + d * (j - 1);
    if (tj >= 0 && tj < TLEN) {
      const size_t off = ((size_t)b * TLEN + tj) * CCH + c0;
      kv[j] = *(const v8bf*)(k + off);
      vv[j] = *(const v8bf*)(v + off);
    } else {
#pragma unroll
      for (int r = 0; r < 8; ++r) { kv[j][r] = f2bf(0.f); vv[j][r] = f2bf(0.f); }
    }
  }
  v8bf outv;
#pragma unroll
  for (int r = 0; r < 8; ++r) {
    const float qf = bf2f(qv[r]);
    float s[3], w[3], mx = -1e30f;
#pragma unroll
    for (int j = 0; j < 3; ++j) { s[j] = qf * bf2f(kv[j][r]); mx = fmaxf(mx, s[j]); }
    float den = 0.f;
#pragma unroll
    for (int j = 0; j < 3; ++j) { w[j] = __expf(s[j] - mx); den += w[j]; }
    const float inv = 1.f / den;
    float o = 0.f;
#pragma unroll
    for (int j = 0; j < 3; ++j)
      o += w[j] * (bf2f(vv[j][r]) + rel[(size_t)(c0 + r) * 3 + j]);
    outv[r] = f2bf(fmaxf(o * inv, 0.f));   // relu
  }
  *(v8bf*)(a + base) = outv;
}

// ---------------------------------------------------------------------------
// Conversions
// ---------------------------------------------------------------------------
__global__ void cvt_f32_to_bf16(const float* __restrict__ src, __bf16* __restrict__ dst, size_t n) {
  for (size_t i = (size_t)blockIdx.x * blockDim.x + threadIdx.x; i < n;
       i += (size_t)gridDim.x * blockDim.x)
    dst[i] = f2bf(src[i]);
}

__global__ void cvt_pad_outw(const float* __restrict__ src, __bf16* __restrict__ dst) {
  const size_t n = (size_t)NCPAD * CCH;
  for (size_t i = (size_t)blockIdx.x * blockDim.x + threadIdx.x; i < n;
       i += (size_t)gridDim.x * blockDim.x) {
    const int r = (int)(i / CCH);
    dst[i] = f2bf(r < NCLS ? src[i] : 0.f);
  }
}

// x (B, DIM, T) f32  ->  xt (B, T, DIM) bf16, LDS-tiled transpose
__global__ __launch_bounds__(256)
void transpose_cvt(const float* __restrict__ x, __bf16* __restrict__ xt) {
  __shared__ float tile[32][33];
  const int b  = blockIdx.z;
  const int t0 = blockIdx.x * 32;
  const int c0 = blockIdx.y * 32;
#pragma unroll
  for (int it = 0; it < 4; ++it) {
    const int r = threadIdx.y + it * 8;
    tile[r][threadIdx.x] = x[((size_t)b * DIMX + c0 + r) * TLEN + t0 + threadIdx.x];
  }
  __syncthreads();
#pragma unroll
  for (int it = 0; it < 4; ++it) {
    const int r = threadIdx.y + it * 8;
    xt[((size_t)b * TLEN + t0 + r) * DIMX + c0 + threadIdx.x] = f2bf(tile[threadIdx.x][r]);
  }
}

// ---------------------------------------------------------------------------
extern "C" void kernel_launch(void* const* d_in, const int* in_sizes, int n_in,
                              void* d_out, int out_size, void* d_ws, size_t ws_size,
                              hipStream_t stream) {
  const float* x     = (const float*)d_in[0];
  const float* mask  = (const float*)d_in[1];
  const float* in_w  = (const float*)d_in[2];
  const float* in_b  = (const float*)d_in[3];
  const float* wq    = (const float*)d_in[4];
  const float* wk    = (const float*)d_in[5];
  const float* wv    = (const float*)d_in[6];
  const float* rel   = (const float*)d_in[7];
  const float* w1    = (const float*)d_in[8];
  const float* b1    = (const float*)d_in[9];
  const float* out_w = (const float*)d_in[10];
  const float* out_b = (const float*)d_in[11];
  float* out = (float*)d_out;

  char* wsp = (char*)d_ws;
  size_t off = 0;
  auto take = [&](size_t bytes) -> char* {
    char* p = wsp + off;
    off += (bytes + 255) & ~(size_t)255;
    return p;
  };
  __bf16* xt      = (__bf16*)take((size_t)BATCH * TLEN * DIMX * 2);
  __bf16* h       = (__bf16*)take((size_t)BATCH * TLEN * CCH * 2);
  __bf16* qb      = (__bf16*)take((size_t)BATCH * TLEN * CCH * 2);
  __bf16* kb      = (__bf16*)take((size_t)BATCH * TLEN * CCH * 2);
  __bf16* vb      = (__bf16*)take((size_t)BATCH * TLEN * CCH * 2);
  __bf16* ab      = (__bf16*)take((size_t)BATCH * TLEN * CCH * 2);
  __bf16* inw_bf  = (__bf16*)take((size_t)CCH * DIMX * 2);
  __bf16* wq_bf   = (__bf16*)take((size_t)LAYERS * CCH * CCH * 2);
  __bf16* wk_bf   = (__bf16*)take((size_t)LAYERS * CCH * CCH * 2);
  __bf16* wv_bf   = (__bf16*)take((size_t)LAYERS * CCH * CCH * 2);
  __bf16* w1_bf   = (__bf16*)take((size_t)LAYERS * CCH * CCH * 2);
  __bf16* outw_bf = (__bf16*)take((size_t)NCPAD * CCH * 2);

  // weight / input conversions
  cvt_f32_to_bf16<<<256, 256, 0, stream>>>(in_w, inw_bf, (size_t)CCH * DIMX);
  cvt_f32_to_bf16<<<512, 256, 0, stream>>>(wq, wq_bf, (size_t)LAYERS * CCH * CCH);
  cvt_f32_to_bf16<<<512, 256, 0, stream>>>(wk, wk_bf, (size_t)LAYERS * CCH * CCH);
  cvt_f32_to_bf16<<<512, 256, 0, stream>>>(wv, wv_bf, (size_t)LAYERS * CCH * CCH);
  cvt_f32_to_bf16<<<512, 256, 0, stream>>>(w1, w1_bf, (size_t)LAYERS * CCH * CCH);
  cvt_pad_outw<<<64, 256, 0, stream>>>(out_w, outw_bf);
  {
    dim3 tb(32, 8), tg(TLEN / 32, DIMX / 32, BATCH);
    transpose_cvt<<<tg, tb, 0, stream>>>(x, xt);
  }

  // input projection: h = in_w @ x + in_b        (M=512, K=1024, N=4096)
  wmma_gemm_kernel<1><<<256, 256, 0, stream>>>(inw_bf, xt, h, nullptr, in_b,
                                               nullptr, CCH, TLEN, DIMX, CCH);

  const int gemmBlocks = (BATCH * (CCH / 32) * (TLEN / 64)) / 8;  // 256
  const int attnBlocks = (BATCH * TLEN * (CCH / 8)) / 256;        // 2048
  for (int i = 0; i < LAYERS; ++i) {
    const int d = 1 << i;
    const size_t wo = (size_t)i * CCH * CCH;
    wmma_gemm_kernel<0><<<gemmBlocks, 256, 0, stream>>>(
        wq_bf + wo, h, qb, nullptr, nullptr, nullptr, CCH, TLEN, CCH, CCH);
    wmma_gemm_kernel<0><<<gemmBlocks, 256, 0, stream>>>(
        wk_bf + wo, h, kb, nullptr, nullptr, nullptr, CCH, TLEN, CCH, CCH);
    wmma_gemm_kernel<0><<<gemmBlocks, 256, 0, stream>>>(
        wv_bf + wo, h, vb, nullptr, nullptr, nullptr, CCH, TLEN, CCH, CCH);
    dal_attn_kernel<<<attnBlocks, 256, 0, stream>>>(
        qb, kb, vb, rel + (size_t)i * CCH * 3, ab, d);
    wmma_gemm_kernel<2><<<gemmBlocks, 256, 0, stream>>>(
        w1_bf + wo, ab, h, nullptr, b1 + (size_t)i * CCH, mask, CCH, TLEN, CCH, CCH);
  }

  // classifier head: out = (out_w @ h + out_b) * mask  (M=160 padded, valid 157)
  const int headBlocks = (BATCH * (NCPAD / 32) * (TLEN / 64)) / 8;  // 80
  wmma_gemm_kernel<3><<<headBlocks, 256, 0, stream>>>(
      outw_bf, h, nullptr, out, out_b, mask, NCPAD, TLEN, CCH, NCLS);
}
